// LeakyRNNModel_77936476553891
// MI455X (gfx1250) — matrix operand order repaired and checked by reading, not compiled, not run on hardware
//
#include <hip/hip_runtime.h>
#include <math.h>

typedef __bf16 bf16_t;
typedef __attribute__((ext_vector_type(16))) __bf16 v16bf;
typedef __attribute__((ext_vector_type(8)))  __bf16 v8bf;
typedef __attribute__((ext_vector_type(8)))  float  v8f;

#define B_    256
#define T_    2048
#define NIN_  256
#define N_    512
#define NOUT_ 64
#define KTOT  (N_ + NIN_)   // 768
#define ALPHA_ 0.1f

#define GB 4      // row groups
#define GN 16     // col groups
#define NWG (GB*GN)          // 64 persistent workgroups
#define TB 64                // rows per WG
#define TN 32                // cols per WG
#define NTHREADS 256         // 8 wave32s

// ---------------------------------------------------------------------------
// Device-wide generation barrier (persistent-kernel scan sync).
// Release fence before arrival, acquire fence after release from spin.
// ---------------------------------------------------------------------------
__device__ __forceinline__ void grid_barrier(unsigned* cnt, unsigned* gen) {
    __threadfence();          // each thread: flush my global stores to device scope
    __syncthreads();
    if (threadIdx.x == 0) {
        unsigned g = __hip_atomic_load(gen, __ATOMIC_RELAXED, __HIP_MEMORY_SCOPE_AGENT);
        unsigned p = __hip_atomic_fetch_add(cnt, 1u, __ATOMIC_ACQ_REL, __HIP_MEMORY_SCOPE_AGENT);
        if (p == NWG - 1u) {
            __hip_atomic_store(cnt, 0u, __ATOMIC_RELAXED, __HIP_MEMORY_SCOPE_AGENT);
            __hip_atomic_fetch_add(gen, 1u, __ATOMIC_RELEASE, __HIP_MEMORY_SCOPE_AGENT);
        } else {
            while (__hip_atomic_load(gen, __ATOMIC_ACQUIRE, __HIP_MEMORY_SCOPE_AGENT) == g) {
                __builtin_amdgcn_s_sleep(2);
            }
        }
    }
    __syncthreads();
    __builtin_amdgcn_fence(__ATOMIC_ACQUIRE, "agent");  // invalidate stale near caches
}

// ---------------------------------------------------------------------------
// Init: zero barrier words and abuf0 (initial tanh(prev)=tanh(0)=0).
// ---------------------------------------------------------------------------
__global__ void rnn_init_kernel(unsigned* bar, bf16_t* abuf0) {
    int i = blockIdx.x * blockDim.x + threadIdx.x;
    if (i < 2) bar[i] = 0u;
    for (int e = i; e < B_ * N_; e += gridDim.x * blockDim.x)
        abuf0[e] = (bf16_t)0.0f;
}

// ---------------------------------------------------------------------------
// Persistent scan kernel: 64 WGs x 256 threads. Each wave owns one 16x16
// output tile for all 2048 steps; per step it accumulates K=768 with
// 24x v_wmma_f32_16x16x32_bf16 (A from global bf16, B from LDS weight tile).
// ---------------------------------------------------------------------------
__global__ __launch_bounds__(NTHREADS, 1)
void rnn_scan_kernel(const float* __restrict__ x,
                     const float* __restrict__ wk,     // [NIN_, N_]
                     const float* __restrict__ rk,     // [N_, N_]
                     const float* __restrict__ bias,   // [N_]
                     const float* __restrict__ dw,     // [N_, NOUT_]
                     const float* __restrict__ db,     // [NOUT_]
                     float* __restrict__ out,          // [B_, NOUT_]
                     bf16_t* __restrict__ abuf0, bf16_t* __restrict__ abuf1,
                     bf16_t* __restrict__ xbf0,  bf16_t* __restrict__ xbf1,
                     float*  __restrict__ sfin,
                     unsigned* bar_cnt, unsigned* bar_gen)
{
    // Combined weight tile, transposed to [n_local][k] so B-fragment loads are
    // contiguous ds_read_b128s: k<512 -> recurrent_kernel, k>=512 -> kernel.
    __shared__ bf16_t wt[TN * KTOT];          // 32*768*2B = 48 KB

    const int wg   = blockIdx.x;
    const int gb   = wg >> 4;                 // / GN
    const int gn   = wg & 15;                 // % GN
    const int tid  = threadIdx.x;
    const int lane = tid & 31;
    const int wave = tid >> 5;                // 0..7
    const int mt   = wave >> 1;               // 0..3 row tile
    const int nt   = wave & 1;                // 0..1 col tile
    const int l16  = lane & 15;
    const int half = lane >> 4;

    // One-time LDS fill (fp32 -> bf16, transposed)
    for (int idx = tid; idx < TN * KTOT; idx += NTHREADS) {
        int nl = idx / KTOT, k = idx % KTOT;
        int n  = gn * TN + nl;
        float v = (k < N_) ? rk[k * N_ + n] : wk[(k - N_) * N_ + n];
        wt[idx] = (bf16_t)v;
    }
    __syncthreads();

    // A-matrix row for this lane (ISA 16-bit A layout: M = lane%16 both halves)
    const int m_a = gb * TB + mt * 16 + l16;
    // C/D layout: col = lane%16, rows = 8*half + r
    const int ncol = gn * TN + nt * 16 + l16;
    const int mrow0 = gb * TB + mt * 16 + 8 * half;

    const bf16_t* bptr = &wt[(nt * 16 + l16) * KTOT];

    float prev[8];
#pragma unroll
    for (int r = 0; r < 8; ++r) prev[r] = 0.0f;
    const float bv = bias[ncol];

    // Prologue: all WGs cooperatively convert x[:,0,:] -> xbf0
    {
        int e0 = wg * 1024 + tid * 4;                 // covers 64*1024 = 65536
        int b  = e0 >> 8, i = e0 & 255;
        const float* p = x + (size_t)b * T_ * NIN_ + i;
        bf16_t* q = xbf0 + e0;
#pragma unroll
        for (int j = 0; j < 4; ++j) q[j] = (bf16_t)p[j];
    }
    grid_barrier(bar_cnt, bar_gen);

    for (int t = 0; t < T_; ++t) {
        const bf16_t* acur  = (t & 1) ? abuf1 : abuf0;
        const bf16_t* xcur  = (t & 1) ? xbf1  : xbf0;
        bf16_t*       anext = (t & 1) ? abuf0 : abuf1;
        bf16_t*       xnext = (t & 1) ? xbf0  : xbf1;

        const bf16_t* arow = acur + (size_t)m_a * N_;
        const bf16_t* xrow = xcur + (size_t)m_a * NIN_;

        v8f c = {};
        // K = 0..511 : tanh(prev) @ recurrent_kernel
#pragma unroll 4
        for (int ks = 0; ks < 16; ++ks) {
            union { v16bf v; v8bf h[2]; } a, b;
            const bf16_t* as = arow + ks * 32 + half * 8;
            a.h[0] = *(const v8bf*)(as);
            a.h[1] = *(const v8bf*)(as + 16);
            const bf16_t* bs = bptr + ks * 32 + half * 8;
            b.h[0] = *(const v8bf*)(bs);
            b.h[1] = *(const v8bf*)(bs + 16);
            c = __builtin_amdgcn_wmma_f32_16x16x32_bf16(false, a.v, false, b.v,
                                                        (short)0, c, false, false);
        }
        // K = 512..767 : x_t @ kernel
#pragma unroll 4
        for (int ks = 0; ks < 8; ++ks) {
            union { v16bf v; v8bf h[2]; } a, b;
            const bf16_t* as = xrow + ks * 32 + half * 8;
            a.h[0] = *(const v8bf*)(as);
            a.h[1] = *(const v8bf*)(as + 16);
            const bf16_t* bs = bptr + N_ + ks * 32 + half * 8;
            b.h[0] = *(const v8bf*)(bs);
            b.h[1] = *(const v8bf*)(bs + 16);
            c = __builtin_amdgcn_wmma_f32_16x16x32_bf16(false, a.v, false, b.v,
                                                        (short)0, c, false, false);
        }

        // Leaky update + activation, all in fp32 registers; publish tanh(out)
        // as bf16 for next step's A matrix.
#pragma unroll
        for (int r = 0; r < 8; ++r) {
            float ns = (1.0f - ALPHA_) * prev[r] + ALPHA_ * (c[r] + bv);
            float o  = tanhf(ns);
            prev[r]  = o;
            anext[(size_t)(mrow0 + r) * N_ + ncol] = (bf16_t)tanhf(o);
        }

        // Prepare x_{t+1} in bf16 one step ahead (spread across all WGs)
        if (t + 1 < T_) {
            int e0 = wg * 1024 + tid * 4;
            int b  = e0 >> 8, i = e0 & 255;
            const float* p = x + (size_t)b * T_ * NIN_ + (size_t)(t + 1) * NIN_ + i;
            bf16_t* q = xnext + e0;
#pragma unroll
            for (int j = 0; j < 4; ++j) q[j] = (bf16_t)p[j];
            if (t + 2 < T_) __builtin_prefetch(p + NIN_, 0, 1);
        }

        grid_barrier(bar_cnt, bar_gen);
    }

    // Publish final state, then compute dense head: out = last @ dw + db
#pragma unroll
    for (int r = 0; r < 8; ++r)
        sfin[(size_t)(mrow0 + r) * N_ + ncol] = prev[r];
    grid_barrier(bar_cnt, bar_gen);

    {
        int idx = wg * NTHREADS + tid;        // 64*256 = 16384 = B_*NOUT_
        int b = idx >> 6, o = idx & 63;
        const float* srow = sfin + (size_t)b * N_;
        float acc = db[o];
        for (int k = 0; k < N_; ++k)
            acc = fmaf(srow[k], dw[k * NOUT_ + o], acc);
        out[idx] = acc;
    }
}

// ---------------------------------------------------------------------------
extern "C" void kernel_launch(void* const* d_in, const int* in_sizes, int n_in,
                              void* d_out, int out_size, void* d_ws, size_t ws_size,
                              hipStream_t stream) {
    (void)in_sizes; (void)n_in; (void)out_size; (void)ws_size;
    const float* x    = (const float*)d_in[0];
    const float* wk   = (const float*)d_in[1];
    const float* rk   = (const float*)d_in[2];
    const float* bias = (const float*)d_in[3];
    const float* dw   = (const float*)d_in[4];
    const float* db   = (const float*)d_in[5];
    float* out = (float*)d_out;

    char* ws = (char*)d_ws;
    unsigned* bar = (unsigned*)ws;                       // 2 words, pad to 256B
    size_t off = 256;
    bf16_t* abuf0 = (bf16_t*)(ws + off); off += (size_t)B_ * N_   * sizeof(bf16_t);
    bf16_t* abuf1 = (bf16_t*)(ws + off); off += (size_t)B_ * N_   * sizeof(bf16_t);
    bf16_t* xbf0  = (bf16_t*)(ws + off); off += (size_t)B_ * NIN_ * sizeof(bf16_t);
    bf16_t* xbf1  = (bf16_t*)(ws + off); off += (size_t)B_ * NIN_ * sizeof(bf16_t);
    float*  sfin  = (float*) (ws + off); off += (size_t)B_ * N_   * sizeof(float);

    hipLaunchKernelGGL(rnn_init_kernel, dim3(128), dim3(256), 0, stream, bar, abuf0);
    hipLaunchKernelGGL(rnn_scan_kernel, dim3(NWG), dim3(NTHREADS), 0, stream,
                       x, wk, rk, bias, dw, db, out,
                       abuf0, abuf1, xbf0, xbf1, sfin, bar, bar + 1);
}
